// AccidentDetection_56152402428224
// MI455X (gfx1250) — matrix operand order repaired and checked by reading, not compiled, not run on hardware
//
#include <hip/hip_runtime.h>
#include <stdint.h>

// ---------------------------------------------------------------------------
// MI455X (gfx1250) implementation.
// Big GEMMs + recurrent LSTM/attention chain, all matmuls via
// v_wmma_f32_16x16x32_bf16 with split-bf16 (hi+lo) for ~f32 accuracy.
// A operands packed pair-major per row, B operands pair-major per column,
// matching the CDNA5 WMMA VGPR layouts (ISA 7.12.2) so the inner loop is
// uint4 global loads -> WMMA, no shuffles.
// ---------------------------------------------------------------------------

typedef __attribute__((ext_vector_type(16))) __bf16 v16bf;
typedef __attribute__((ext_vector_type(8)))  float  v8f;

union BfOp { v16bf v; uint4 q[2]; };

__device__ __forceinline__ uint32_t f2bf(float f) {
  union { float f; uint32_t u; } v; v.f = f;
  uint32_t x = v.u;
  uint32_t r = x + 0x7FFFu + ((x >> 16) & 1u);   // round-to-nearest-even
  return r >> 16;
}
__device__ __forceinline__ float bf2f(uint32_t h) {
  union { uint32_t u; float f; } v; v.u = h << 16;
  return v.f;
}
__device__ __forceinline__ float sigf(float x) { return 1.0f / (1.0f + expf(-x)); }

// ---------------------------------------------------------------------------
// Pack f32 matrix into split-bf16 pair-major planes.
// dst[outer * (Kd/2) + p] = bf16(x[2p]) | bf16(x[2p+1])<<16   (hi plane)
// dst[planeOff + ...]     = same for the lo residual.
// For A operands: outer = row.  For B operands: outer = column.
// ---------------------------------------------------------------------------
__global__ void pack_pairs_kernel(const float* __restrict__ src,
                                  long long stride0, long long strideK,
                                  uint32_t* __restrict__ dst, long long planeOff,
                                  int nOuter, int Kd)
{
  int pairs = Kd >> 1;
  long long np = (long long)nOuter * pairs;
  long long i = (long long)blockIdx.x * blockDim.x + threadIdx.x;
  if (i >= np) return;
  long long outer = i / pairs;
  int p = (int)(i - outer * pairs);
  const float* s = src + outer * stride0 + (long long)(2 * p) * strideK;
  float x0 = s[0], x1 = s[strideK];
  uint32_t h0 = f2bf(x0), h1 = f2bf(x1);
  float l0 = x0 - bf2f(h0), l1 = x1 - bf2f(h1);
  dst[i]            = h0 | (h1 << 16);
  dst[i + planeOff] = f2bf(l0) | (f2bf(l1) << 16);
}

// ---------------------------------------------------------------------------
// Generic WMMA GEMM:  C[M,N] (f32) = mapRows(A) @ B (+bias) (+C if acc)
// A packed split-bf16 pair-major per row (row stride aRowStride dwords,
// lo plane at +aPlane), B packed split-bf16 pair-major per column
// (column stride = Kd/2 dwords, lo plane at +bPlane).
// Block = 128 threads = 4 waves; block tile = 16 rows x 256 cols
// (each wave: 16x64 = 4 WMMA n-tiles). grid = (N/256, M/16).
// rowMode: 0 identity, 1 img rows (r -> r*20), 2 obj rows.
// 3 WMMAs per tile per 32-K step: ah*bh + al*bh + ah*bl  (~f32 precision).
// ---------------------------------------------------------------------------
#define GEMM_NT 4
__launch_bounds__(128)
__global__ void gemm_wmma_kernel(const uint32_t* __restrict__ Apk, int aRowStride,
                                 long long aPlane,
                                 const uint32_t* __restrict__ Bpk, long long bPlane,
                                 const float* __restrict__ bias,
                                 float* __restrict__ C, int N, int Kd,
                                 int acc, int rowMode)
{
  const int wave = threadIdx.x >> 5;
  const int lane = threadIdx.x & 31;
  const int hi   = (lane >> 4) & 1;
  const int l15  = lane & 15;
  const int mtile = blockIdx.y;
  const int ncol0 = (blockIdx.x * 4 + wave) * (GEMM_NT * 16);

  // map logical row -> physical packed row of A
  int lr = mtile * 16 + l15;
  int prow;
  if (rowMode == 1)      prow = lr * 20;                         // image slot
  else if (rowMode == 2) prow = (lr / 19) * 20 + 1 + (lr % 19);  // object slots
  else                   prow = lr;

  const uint32_t* aHi = Apk + (size_t)prow * aRowStride;
  const uint32_t* aLo = aHi + aPlane;

  v8f accv[GEMM_NT];
#pragma unroll
  for (int nt = 0; nt < GEMM_NT; nt++)
#pragma unroll
    for (int r = 0; r < 8; r++) accv[nt][r] = 0.0f;

  const int npairs = Kd >> 1;
  const int aoff0 = hi ? 4 : 0;   // A: lanes>=16 take K+8 (pairs +4)
  const int boff0 = hi ? 4 : 0;   // B: lanes>=16 take K+8 (pairs +4)

  for (int pb = 0; pb < npairs; pb += 16) {     // 32 K values per step
    BfOp ah, al;
    ah.q[0] = *(const uint4*)(aHi + pb + aoff0);
    ah.q[1] = *(const uint4*)(aHi + pb + 8 + aoff0);
    al.q[0] = *(const uint4*)(aLo + pb + aoff0);
    al.q[1] = *(const uint4*)(aLo + pb + 8 + aoff0);
#pragma unroll
    for (int nt = 0; nt < GEMM_NT; nt++) {
      const int col = ncol0 + nt * 16 + l15;
      const uint32_t* bH = Bpk + (size_t)col * npairs + pb + boff0;
      BfOp bh, bl;
      bh.q[0] = *(const uint4*)(bH);
      bh.q[1] = *(const uint4*)(bH + 8);
      bl.q[0] = *(const uint4*)(bH + bPlane);
      bl.q[1] = *(const uint4*)(bH + bPlane + 8);
      accv[nt] = __builtin_amdgcn_wmma_f32_16x16x32_bf16(
          false, ah.v, false, bh.v, (short)0, accv[nt], false, false);
      accv[nt] = __builtin_amdgcn_wmma_f32_16x16x32_bf16(
          false, al.v, false, bh.v, (short)0, accv[nt], false, false);
      accv[nt] = __builtin_amdgcn_wmma_f32_16x16x32_bf16(
          false, ah.v, false, bl.v, (short)0, accv[nt], false, false);
    }
  }

  // epilogue: C layout per ISA: lane 0-15 -> M=r, lane 16-31 -> M=8+r; N=lane&15
#pragma unroll
  for (int nt = 0; nt < GEMM_NT; nt++) {
    const int col = ncol0 + nt * 16 + l15;
    const float b = bias ? bias[col] : 0.0f;
#pragma unroll
    for (int r = 0; r < 8; r++) {
      const int row = mtile * 16 + r + (hi ? 8 : 0);
      const size_t idx = (size_t)row * N + col;
      float v = accv[nt][r] + b;
      if (acc) v += C[idx];
      C[idx] = v;
    }
  }
}

// ---------------------------------------------------------------------------
// obj_mask[r] = sum_d x[phys(r)*4096 + d],  r in [0,19456)
// ---------------------------------------------------------------------------
__global__ void mask_kernel(const float* __restrict__ x, float* __restrict__ mask)
{
  const int r = blockIdx.x;
  const int phys = (r / 19) * 20 + 1 + (r % 19);
  const float* row = x + (size_t)phys * 4096;
  float s = 0.0f;
  for (int d = threadIdx.x; d < 4096; d += 256) s += row[d];
  __shared__ float red[256];
  red[threadIdx.x] = s;
  __syncthreads();
  for (int st = 128; st > 0; st >>= 1) {
    if (threadIdx.x < st) red[threadIdx.x] += red[threadIdx.x + st];
    __syncthreads();
  }
  if (threadIdx.x == 0) mask[r] = red[0];
}

// ---------------------------------------------------------------------------
// objpk = pack(obj_feat * mask) : 19456 rows x 1024 -> split-bf16 pairs
// ---------------------------------------------------------------------------
__global__ void mask_pack_kernel(const float* __restrict__ feat,
                                 const float* __restrict__ mask,
                                 uint32_t* __restrict__ dst, long long planeOff)
{
  long long i = (long long)blockIdx.x * blockDim.x + threadIdx.x;
  if (i >= (long long)19456 * 512) return;
  const int r = (int)(i >> 9);
  const int p = (int)(i & 511);
  const float m = mask[r];
  const float* s = feat + (size_t)r * 1024 + 2 * p;
  float x0 = s[0] * m, x1 = s[1] * m;
  uint32_t h0 = f2bf(x0), h1 = f2bf(x1);
  float l0 = x0 - bf2f(h0), l1 = x1 - bf2f(h1);
  dst[i]            = h0 | (h1 << 16);
  dst[i + planeOff] = f2bf(l0) | (f2bf(l1) << 16);
}

// ---------------------------------------------------------------------------
// Attention step: scores -> softmax -> alphas (to d_out) -> fusion buffer.
// One block per batch row b (16 blocks, 256 threads = 8 waves).
// ---------------------------------------------------------------------------
__global__ void attn_kernel(const float* __restrict__ hW,
                            const float* __restrict__ a,
                            const float* __restrict__ mask,
                            const float* __restrict__ Wj,
                            const float* __restrict__ img_feat,
                            float* __restrict__ fusion,
                            float* __restrict__ alphas_out, int t)
{
  const int b = blockIdx.x;
  const int wv = threadIdx.x >> 5, lane = threadIdx.x & 31;
  __shared__ float sc[32];
  __shared__ float al[32];

  const long long rb = ((long long)(b * 64 + t)) * 19;
  const float* hrow = hW + b * 1024;

  for (int j = wv; j < 19; j += 8) {
    const float* arow = a + (rb + j) * 1024;
    float s = 0.0f;
    for (int d = lane; d < 1024; d += 32)
      s += tanhf(hrow[d] + arow[d]) * Wj[d];
    for (int o = 16; o > 0; o >>= 1) s += __shfl_down(s, o, 32);
    if (lane == 0) sc[j] = s * mask[rb + j];
  }
  __syncthreads();

  if (threadIdx.x < 32) {
    float v = (threadIdx.x < 19) ? sc[threadIdx.x] : -1e30f;
    float m = v;
    for (int o = 16; o > 0; o >>= 1) m = fmaxf(m, __shfl_xor(m, o, 32));
    float e = (threadIdx.x < 19) ? expf(v - m) : 0.0f;
    float ss = e;
    for (int o = 16; o > 0; o >>= 1) ss += __shfl_xor(ss, o, 32);
    if (threadIdx.x < 19) {
      float alpha = e / ss;
      al[threadIdx.x] = alpha;
      alphas_out[(long long)t * 16 * 19 + b * 19 + threadIdx.x] = alpha;
    }
  }
  __syncthreads();

  for (int d = threadIdx.x; d < 1024; d += 256) {
    float ws = 0.0f;
#pragma unroll
    for (int j = 0; j < 19; j++) ws += a[(rb + j) * 1024 + d] * al[j];
    fusion[b * 2048 + 1024 + d] = ws;
    fusion[b * 2048 + d] = img_feat[((long long)(b * 64 + t)) * 1024 + d];
  }
}

// ---------------------------------------------------------------------------
// LSTM gate nonlinearity: gates (16x4096, i|f|g|o) + c -> h,c in place.
// ---------------------------------------------------------------------------
__global__ void lstm_update_kernel(const float* __restrict__ gates,
                                   float* __restrict__ h, float* __restrict__ c)
{
  const int i = blockIdx.x * 256 + threadIdx.x;
  if (i >= 16 * 1024) return;
  const int b = i >> 10, d = i & 1023;
  const float* g = gates + (size_t)b * 4096;
  const float ig = sigf(g[d]);
  const float fg = sigf(g[1024 + d]);
  const float gg = tanhf(g[2048 + d]);
  const float og = sigf(g[3072 + d]);
  const float cn = fg * c[i] + ig * gg;
  c[i] = cn;
  h[i] = og * tanhf(cn);
}

// ---------------------------------------------------------------------------
// pred[t,b] = sigmoid(h1[b,:] . W_out + b_out)
// ---------------------------------------------------------------------------
__global__ void pred_kernel(const float* __restrict__ h,
                            const float* __restrict__ Wout,
                            const float* __restrict__ bout,
                            float* __restrict__ out, int t)
{
  const int b = blockIdx.x;
  float s = 0.0f;
  for (int d = threadIdx.x; d < 1024; d += 256) s += h[b * 1024 + d] * Wout[d];
  __shared__ float red[256];
  red[threadIdx.x] = s;
  __syncthreads();
  for (int st = 128; st > 0; st >>= 1) {
    if (threadIdx.x < st) red[threadIdx.x] += red[threadIdx.x + st];
    __syncthreads();
  }
  if (threadIdx.x == 0)
    out[19456 + t * 16 + b] = sigf(red[0] + bout[0]);
}

__global__ void zero_kernel(float* __restrict__ p, long long n)
{
  long long i = (long long)blockIdx.x * blockDim.x + threadIdx.x;
  if (i < n) p[i] = 0.0f;
}

// ---------------------------------------------------------------------------
// Host driver
// ---------------------------------------------------------------------------
static inline size_t alignup(size_t x) { return (x + 255) & ~(size_t)255; }

extern "C" void kernel_launch(void* const* d_in, const int* in_sizes, int n_in,
                              void* d_out, int out_size, void* d_ws, size_t ws_size,
                              hipStream_t stream)
{
  (void)in_sizes; (void)n_in; (void)out_size; (void)ws_size;
  const float* x      = (const float*)d_in[0];
  const float* W_img  = (const float*)d_in[1];
  const float* b_img  = (const float*)d_in[2];
  const float* W_obj  = (const float*)d_in[3];
  const float* b_obj  = (const float*)d_in[4];
  const float* W_obj2 = (const float*)d_in[5];
  const float* b_obj2 = (const float*)d_in[6];
  const float* W_ah   = (const float*)d_in[7];
  const float* W_aj   = (const float*)d_in[8];
  const float* W_ih0  = (const float*)d_in[9];
  const float* W_hh0  = (const float*)d_in[10];
  const float* b0     = (const float*)d_in[11];
  const float* W_ih1  = (const float*)d_in[12];
  const float* W_hh1  = (const float*)d_in[13];
  const float* b1     = (const float*)d_in[14];
  const float* W_out  = (const float*)d_in[15];
  const float* b_out  = (const float*)d_in[16];
  float* out = (float*)d_out;

  // ---- workspace layout (bytes) ----
  char* ws = (char*)d_ws;
  size_t off = 0;
  auto take = [&](size_t bytes) { size_t o = off; off = alignup(off + bytes); return o; };
  const size_t XPK   = take(2ull * 20480 * 2048 * 4);   // x split-bf16 pairs
  const size_t WIMG  = take(2ull * 1024 * 2048 * 4);
  const size_t WOBJ  = take(2ull * 1024 * 2048 * 4);
  const size_t WOBJ2 = take(2ull * 1024 * 512 * 4);
  const size_t WATTN = take(2ull * 1024 * 512 * 4);
  const size_t WIH0  = take(2ull * 4096 * 1024 * 4);
  const size_t WHH0  = take(2ull * 4096 * 512 * 4);
  const size_t WIH1  = take(2ull * 4096 * 512 * 4);
  const size_t WHH1  = take(2ull * 4096 * 512 * 4);
  const size_t OBJF  = take(19456ull * 1024 * 4);       // obj_feat, later reused as 'a'
  const size_t OBJPK = take(2ull * 19456 * 512 * 4);
  const size_t IMGF  = take(1024ull * 1024 * 4);
  const size_t MASK  = take(19456ull * 4);
  const size_t STATE = take(4ull * 16 * 1024 * 4);      // h0,c0,h1,c1 contiguous
  const size_t H1PK  = take(2ull * 16 * 512 * 4);
  const size_t H0PK  = take(2ull * 16 * 512 * 4);
  const size_t FUSPK = take(2ull * 16 * 1024 * 4);
  const size_t HWB   = take(16ull * 1024 * 4);
  const size_t GATES = take(16ull * 4096 * 4);
  const size_t FUS   = take(16ull * 2048 * 4);

  uint32_t* xpk   = (uint32_t*)(ws + XPK);
  uint32_t* wimg  = (uint32_t*)(ws + WIMG);
  uint32_t* wobj  = (uint32_t*)(ws + WOBJ);
  uint32_t* wobj2 = (uint32_t*)(ws + WOBJ2);
  uint32_t* wattn = (uint32_t*)(ws + WATTN);
  uint32_t* wih0  = (uint32_t*)(ws + WIH0);
  uint32_t* whh0  = (uint32_t*)(ws + WHH0);
  uint32_t* wih1  = (uint32_t*)(ws + WIH1);
  uint32_t* whh1  = (uint32_t*)(ws + WHH1);
  float*    objf  = (float*)(ws + OBJF);
  float*    abuf  = objf;                       // alias: obj_feat consumed before 'a' written
  uint32_t* objpk = (uint32_t*)(ws + OBJPK);
  float*    imgf  = (float*)(ws + IMGF);
  float*    maskb = (float*)(ws + MASK);
  float*    h0    = (float*)(ws + STATE);
  float*    c0    = h0 + 16 * 1024;
  float*    h1    = h0 + 2 * 16 * 1024;
  float*    c1    = h0 + 3 * 16 * 1024;
  uint32_t* h1pk  = (uint32_t*)(ws + H1PK);
  uint32_t* h0pk  = (uint32_t*)(ws + H0PK);
  uint32_t* fuspk = (uint32_t*)(ws + FUSPK);
  float*    hWb   = (float*)(ws + HWB);
  float*    gates = (float*)(ws + GATES);
  float*    fus   = (float*)(ws + FUS);

  auto packGrid = [](long long nOuter, int Kd) {
    long long np = nOuter * (Kd >> 1);
    return dim3((unsigned)((np + 255) / 256));
  };

  // ---- pack x (all 20480 rows of length 4096) and all weights ----
  pack_pairs_kernel<<<packGrid(20480, 4096), 256, 0, stream>>>(
      x, 4096, 1, xpk, 20480ll * 2048, 20480, 4096);
  // W as B operands: outer = output column n
  pack_pairs_kernel<<<packGrid(1024, 4096), 256, 0, stream>>>(
      W_img, 1, 1024, wimg, 1024ll * 2048, 1024, 4096);
  pack_pairs_kernel<<<packGrid(1024, 4096), 256, 0, stream>>>(
      W_obj, 1, 1024, wobj, 1024ll * 2048, 1024, 4096);
  pack_pairs_kernel<<<packGrid(1024, 1024), 256, 0, stream>>>(
      W_obj2, 1, 1024, wobj2, 1024ll * 512, 1024, 1024);
  pack_pairs_kernel<<<packGrid(1024, 1024), 256, 0, stream>>>(
      W_ah, 1, 1024, wattn, 1024ll * 512, 1024, 1024);
  // LSTM weights are used transposed: B[k][n] = W[n*K + k]
  pack_pairs_kernel<<<packGrid(4096, 2048), 256, 0, stream>>>(
      W_ih0, 2048, 1, wih0, 4096ll * 1024, 4096, 2048);
  pack_pairs_kernel<<<packGrid(4096, 1024), 256, 0, stream>>>(
      W_hh0, 1024, 1, whh0, 4096ll * 512, 4096, 1024);
  pack_pairs_kernel<<<packGrid(4096, 1024), 256, 0, stream>>>(
      W_ih1, 1024, 1, wih1, 4096ll * 512, 4096, 1024);
  pack_pairs_kernel<<<packGrid(4096, 1024), 256, 0, stream>>>(
      W_hh1, 1024, 1, whh1, 4096ll * 512, 4096, 1024);

  // ---- obj_mask ----
  mask_kernel<<<19456, 256, 0, stream>>>(x, maskb);

  // ---- img_feat = x[:,:,0,:] @ W_img + b_img   (M=1024) ----
  gemm_wmma_kernel<<<dim3(4, 64), 128, 0, stream>>>(
      xpk, 2048, 20480ll * 2048, wimg, 1024ll * 2048, b_img, imgf, 1024, 4096, 0, 1);

  // ---- obj_feat = x[:,:,1:,:] @ W_obj + b_obj  (M=19456) ----
  gemm_wmma_kernel<<<dim3(4, 1216), 128, 0, stream>>>(
      xpk, 2048, 20480ll * 2048, wobj, 1024ll * 2048, b_obj, objf, 1024, 4096, 0, 2);

  // ---- mask & repack obj_feat ----
  mask_pack_kernel<<<(unsigned)((19456ll * 512 + 255) / 256), 256, 0, stream>>>(
      objf, maskb, objpk, 19456ll * 512);

  // ---- a = obj_feat_masked @ W_obj2 + b_obj2  (writes over objf alias) ----
  gemm_wmma_kernel<<<dim3(4, 1216), 128, 0, stream>>>(
      objpk, 512, 19456ll * 512, wobj2, 1024ll * 512, b_obj2, abuf, 1024, 1024, 0, 0);

  // ---- init LSTM state ----
  zero_kernel<<<(4 * 16 * 1024 + 255) / 256, 256, 0, stream>>>(h0, 4ll * 16 * 1024);

  // ---- recurrent scan over 64 frames ----
  for (int t = 0; t < 64; t++) {
    // pack h1_prev (used by attention GEMM and by layer-1 recurrent GEMM)
    pack_pairs_kernel<<<packGrid(16, 1024), 256, 0, stream>>>(
        h1, 1024, 1, h1pk, 16ll * 512, 16, 1024);
    // hW = h1_prev @ W_attn_h
    gemm_wmma_kernel<<<dim3(4, 1), 128, 0, stream>>>(
        h1pk, 512, 16ll * 512, wattn, 1024ll * 512, nullptr, hWb, 1024, 1024, 0, 0);
    // attention -> alphas (d_out) + fusion
    attn_kernel<<<16, 256, 0, stream>>>(hWb, abuf, maskb, W_aj, imgf, fus, out, t);
    // layer 0
    pack_pairs_kernel<<<packGrid(16, 2048), 256, 0, stream>>>(
        fus, 2048, 1, fuspk, 16ll * 1024, 16, 2048);
    pack_pairs_kernel<<<packGrid(16, 1024), 256, 0, stream>>>(
        h0, 1024, 1, h0pk, 16ll * 512, 16, 1024);
    gemm_wmma_kernel<<<dim3(16, 1), 128, 0, stream>>>(
        fuspk, 1024, 16ll * 1024, wih0, 4096ll * 1024, b0, gates, 4096, 2048, 0, 0);
    gemm_wmma_kernel<<<dim3(16, 1), 128, 0, stream>>>(
        h0pk, 512, 16ll * 512, whh0, 4096ll * 512, nullptr, gates, 4096, 1024, 1, 0);
    lstm_update_kernel<<<64, 256, 0, stream>>>(gates, h0, c0);
    // layer 1
    pack_pairs_kernel<<<packGrid(16, 1024), 256, 0, stream>>>(
        h0, 1024, 1, h0pk, 16ll * 512, 16, 1024);
    gemm_wmma_kernel<<<dim3(16, 1), 128, 0, stream>>>(
        h0pk, 512, 16ll * 512, wih1, 4096ll * 512, b1, gates, 4096, 1024, 0, 0);
    gemm_wmma_kernel<<<dim3(16, 1), 128, 0, stream>>>(
        h1pk, 512, 16ll * 512, whh1, 4096ll * 512, nullptr, gates, 4096, 1024, 1, 0);
    lstm_update_kernel<<<64, 256, 0, stream>>>(gates, h1, c1);
    // prediction
    pred_kernel<<<16, 256, 0, stream>>>(h1, W_out, b_out, out, t);
  }
}
